// BottleneckR_30305289240891
// MI455X (gfx1250) — matrix operand order repaired and checked by reading, not compiled, not run on hardware
//
#include <hip/hip_runtime.h>
#include <hip/hip_bf16.h>

typedef __bf16 bf16_t;
typedef __attribute__((ext_vector_type(16))) __bf16 v16bf;
typedef __attribute__((ext_vector_type(8)))  __bf16 v8bf;
typedef __attribute__((ext_vector_type(4)))  __bf16 v4bf;
typedef __attribute__((ext_vector_type(8)))  float  v8f;
typedef int int4v __attribute__((ext_vector_type(4)));
typedef __attribute__((address_space(1))) int4v g_int4;   // global
typedef __attribute__((address_space(3))) int4v l_int4;   // LDS

// CDNA5 async global->LDS staging (ASYNCcnt path), guarded so a toolchain
// without the builtins falls back to load+ds_store.
#if defined(__has_builtin)
#if __has_builtin(__builtin_amdgcn_global_load_async_to_lds_b128) && \
    __has_builtin(__builtin_amdgcn_s_wait_asynccnt)
#define USE_ASYNC_LDS 1
#endif
#endif

// ---------------------------------------------------------------------------
// Per-channel BN statistics -> folded scale/shift:  y = x*s + t
// One block per channel; NCHW layout; no integer division in the hot loop.
// ---------------------------------------------------------------------------
__global__ __launch_bounds__(256)
void bn_stats_kernel(const float* __restrict__ x,
                     const float* __restrict__ gamma,
                     const float* __restrict__ beta,
                     float* __restrict__ s, float* __restrict__ t,
                     int N, int C, int HW) {
  __shared__ float ssum[256];
  __shared__ float ssq[256];
  const int c = blockIdx.x;
  const int tid = threadIdx.x;
  float sum = 0.f, sq = 0.f;
  for (int n = 0; n < N; ++n) {
    const float* xr = x + ((long)n * C + c) * HW;
    for (int hw = tid; hw < HW; hw += 256) {
      float v = xr[hw];
      sum += v;
      sq  += v * v;
    }
  }
  ssum[tid] = sum; ssq[tid] = sq;
  __syncthreads();
  for (int off = 128; off > 0; off >>= 1) {
    if (tid < off) { ssum[tid] += ssum[tid + off]; ssq[tid] += ssq[tid + off]; }
    __syncthreads();
  }
  if (tid == 0) {
    float cnt  = (float)N * (float)HW;
    float mean = ssum[0] / cnt;
    float var  = ssq[0] / cnt - mean * mean;
    float sc   = gamma[c] * rsqrtf(var + 1e-5f);
    s[c] = sc;
    t[c] = beta[c] - mean * sc;
  }
}

// ---------------------------------------------------------------------------
// TBN ternarization per output filter: {-alpha, 0, +alpha} -> bf16
// ---------------------------------------------------------------------------
__global__ __launch_bounds__(256)
void ternarize_kernel(const float* __restrict__ w,
                      bf16_t* __restrict__ wq, int K) {
  __shared__ float red1[256];
  __shared__ float red2[256];
  __shared__ float s_delta, s_alpha;
  const int co = blockIdx.x;
  const int tid = threadIdx.x;
  const float* wr = w + (long)co * K;

  float asum = 0.f;
  for (int k = tid; k < K; k += 256) asum += fabsf(wr[k]);
  red1[tid] = asum;
  __syncthreads();
  for (int off = 128; off > 0; off >>= 1) {
    if (tid < off) red1[tid] += red1[tid + off];
    __syncthreads();
  }
  if (tid == 0) s_delta = 0.7f * red1[0] / (float)K;
  __syncthreads();
  const float delta = s_delta;

  float msum = 0.f, mcnt = 0.f;
  for (int k = tid; k < K; k += 256) {
    float a = fabsf(wr[k]);
    if (a > delta) { msum += a; mcnt += 1.f; }
  }
  red1[tid] = msum; red2[tid] = mcnt;
  __syncthreads();
  for (int off = 128; off > 0; off >>= 1) {
    if (tid < off) { red1[tid] += red1[tid + off]; red2[tid] += red2[tid + off]; }
    __syncthreads();
  }
  if (tid == 0) s_alpha = red1[0] / (red2[0] + 1e-8f);
  __syncthreads();
  const float alpha = s_alpha;

  for (int k = tid; k < K; k += 256) {
    float a = wr[k];
    float q = (fabsf(a) > delta) ? (a > 0.f ? alpha : -alpha) : 0.f;
    wq[(long)co * K + k] = (bf16_t)q;
  }
}

// ---------------------------------------------------------------------------
// Implicit-GEMM convolution, folded-BN input, bf16 WMMA, fp32 accumulate.
// Block: 256 threads = 8 waves; block tile 128(M=co) x 32(N=pixels), BK=32.
// Wave w owns M sub-tile w (16 rows) and computes BOTH 16-wide N sub-tiles,
// reusing its A fragment -> 2 x v_wmma_f32_16x16x32_bf16 per chunk per wave.
// All conv geometry is compile-time -> no integer division in codegen.
// ---------------------------------------------------------------------------
template <int KH, int KW, int STRIDE, int PAD, int HOUT, int WOUT, bool HASBN>
__global__ __launch_bounds__(256)
void conv_wmma_kernel(const float* __restrict__ in,
                      const float* __restrict__ s,
                      const float* __restrict__ tS,
                      const bf16_t* __restrict__ wq,
                      float* __restrict__ out,
                      int Nb, int Cin, int Hin, int Win, int Cout) {
  constexpr int HWo = HOUT * WOUT;
  constexpr int KHW = KH * KW;
  __shared__ bf16_t Asub[128][40];   // [co][kk], stride 40 bf16 (80 B) de-banks
  __shared__ bf16_t Bsub[32][40];    // [pixel][kk]

  const int tid  = threadIdx.x;
  const int lane = tid & 31;
  const int wid  = tid >> 5;          // 0..7  -> M sub-tile
  const int co0  = blockIdx.y * 128;
  const int p0   = blockIdx.x * 32;
  const int Ktot = Cin * KHW;

  // B-tile gather: 8 threads per pixel, 4 consecutive k each (32x32 tile)
  const int nnl = tid >> 3;           // 0..31 local pixel
  const int kk0 = (tid & 7) * 4;      // 0,4,...,28
  const int pB  = p0 + nnl;
  const int nB  = pB / HWo;           // compile-time divisor
  const int qB  = pB - nB * HWo;
  const int ohB = qB / WOUT;
  const int owB = qB - ohB * WOUT;

  // A-tile staging: 128x32 bf16 = 8 KB; each thread moves 32 B (2 x b128)
  const int arow = tid >> 1;          // 0..127
  const int aseg = (tid & 1) * 16;    // 0 or 16

  v8f acc0 = {};
  v8f acc1 = {};

  const int nchunks = Ktot >> 5;
  for (int kc = 0; kc < nchunks; ++kc) {
    const int kbase = kc << 5;

    // --- stage A (ternary bf16 weights) into LDS
    {
      const bf16_t* gsrc = wq + (long)(co0 + arow) * Ktot + kbase + aseg;
#if defined(USE_ASYNC_LDS)
      // builtin signature: (int4 AS(1)*, int4 AS(3)*, imm offset, imm cpol)
      __builtin_amdgcn_global_load_async_to_lds_b128(
          (g_int4*)(int4v*)gsrc,
          (l_int4*)(int4v*)&Asub[arow][aseg], 0, 0);
      __builtin_amdgcn_global_load_async_to_lds_b128(
          (g_int4*)(int4v*)(gsrc + 8),
          (l_int4*)(int4v*)&Asub[arow][aseg + 8], 0, 0);
#else
      *(v8bf*)&Asub[arow][aseg]     = *(const v8bf*)gsrc;
      *(v8bf*)&Asub[arow][aseg + 8] = *(const v8bf*)(gsrc + 8);
#endif
      if (kc + 1 < nchunks)  // global_prefetch_b8 of next weight chunk
        __builtin_prefetch(gsrc + 32, 0, 0);
    }

    // --- stage B: gather 32x32 activations, fold BN, convert to bf16
    {
      v4bf bvals;
#pragma unroll
      for (int j = 0; j < 4; ++j) {
        int k  = kbase + kk0 + j;
        int ci = k / KHW;               // folds to k when KHW==1
        int r  = k - ci * KHW;
        int kh = r / KW;
        int kw = r - kh * KW;
        int ih = ohB * STRIDE - PAD + kh;
        int iw = owB * STRIDE - PAD + kw;
        float v = 0.f;
        if (ih >= 0 && ih < Hin && iw >= 0 && iw < Win) {
          v = in[(((long)nB * Cin + ci) * Hin + ih) * Win + iw];
          if (HASBN) v = v * s[ci] + tS[ci];
        }
        bvals[j] = (bf16_t)v;
      }
      *(v4bf*)&Bsub[nnl][kk0] = bvals;
    }
#if defined(USE_ASYNC_LDS)
    __builtin_amdgcn_s_wait_asynccnt(0);
#endif
    __syncthreads();

    // --- fragments per ISA wave32 VGPR layouts
    // A 16x32 bf16: lanes<16 K {0..7,16..23}; lanes>=16 K {8..15,24..31}
    const int m     = lane & 15;
    const int abase = (lane >= 16) ? 8 : 0;
    v8bf alo = *(const v8bf*)&Asub[wid * 16 + m][abase];
    v8bf ahi = *(const v8bf*)&Asub[wid * 16 + m][abase + 16];
    v16bf afrag = __builtin_shufflevector(alo, ahi,
        0, 1, 2, 3, 4, 5, 6, 7, 8, 9, 10, 11, 12, 13, 14, 15);
    // B 32x16 bf16: lanes<16 K {0..7,16..23}; lanes>=16 K {8..15,24..31}
    const int bbase = (lane >= 16) ? 8 : 0;
    v8bf b0lo = *(const v8bf*)&Bsub[m][bbase];
    v8bf b0hi = *(const v8bf*)&Bsub[m][bbase + 16];
    v16bf bfrag0 = __builtin_shufflevector(b0lo, b0hi,
        0, 1, 2, 3, 4, 5, 6, 7, 8, 9, 10, 11, 12, 13, 14, 15);
    v8bf b1lo = *(const v8bf*)&Bsub[16 + m][bbase];
    v8bf b1hi = *(const v8bf*)&Bsub[16 + m][bbase + 16];
    v16bf bfrag1 = __builtin_shufflevector(b1lo, b1hi,
        0, 1, 2, 3, 4, 5, 6, 7, 8, 9, 10, 11, 12, 13, 14, 15);

    acc0 = __builtin_amdgcn_wmma_f32_16x16x32_bf16(
        false, afrag, false, bfrag0, (short)0, acc0, false, false);
    acc1 = __builtin_amdgcn_wmma_f32_16x16x32_bf16(
        false, afrag, false, bfrag1, (short)0, acc1, false, false);
    __syncthreads();
  }

  // --- write D: lanes<16 -> M=v, lanes>=16 -> M=8+v; N = lane%16
  const int m0 = (lane >= 16) ? 8 : 0;
#pragma unroll
  for (int wn = 0; wn < 2; ++wn) {
    const v8f& a = wn ? acc1 : acc0;
    int nn = wn * 16 + (lane & 15);
    int p  = p0 + nn;
    int n  = p / HWo;                  // compile-time divisor
    int q  = p - n * HWo;
#pragma unroll
    for (int v = 0; v < 8; ++v) {
      int co = co0 + wid * 16 + m0 + v;
      out[((long)n * Cout + co) * HWo + q] = a[v];
    }
  }
}

// ---------------------------------------------------------------------------
// out += BN(shortcut).  One block per (n,c) image plane; C=1024 (pow2 mask).
// ---------------------------------------------------------------------------
__global__ __launch_bounds__(256)
void residual_add_kernel(float* __restrict__ out,
                         const float* __restrict__ sc,
                         const float* __restrict__ s,
                         const float* __restrict__ t,
                         int HW) {
  const int img = blockIdx.x;          // n*C + c
  const int c   = img & 1023;          // C == 1024
  const long base = (long)img * HW;
  const int i = threadIdx.x;
  if (i < HW) out[base + i] += sc[base + i] * s[c] + t[c];
}

// ---------------------------------------------------------------------------
extern "C" void kernel_launch(void* const* d_in, const int* in_sizes, int n_in,
                              void* d_out, int out_size, void* d_ws, size_t ws_size,
                              hipStream_t stream) {
  (void)in_sizes; (void)n_in; (void)out_size; (void)ws_size;

  const float* x      = (const float*)d_in[0];
  const float* bn1_g  = (const float*)d_in[1];
  const float* bn1_b  = (const float*)d_in[2];
  const float* w1     = (const float*)d_in[3];
  const float* bn2_g  = (const float*)d_in[4];
  const float* bn2_b  = (const float*)d_in[5];
  const float* w2     = (const float*)d_in[6];
  const float* bn3_g  = (const float*)d_in[7];
  const float* bn3_b  = (const float*)d_in[8];
  const float* w3     = (const float*)d_in[9];
  const float* ds_w   = (const float*)d_in[10];
  const float* ds_g   = (const float*)d_in[11];
  const float* ds_b   = (const float*)d_in[12];
  float* out = (float*)d_out;

  const int N = 64, C1 = 512, H = 28, W = 28;
  const int P = 256;            // planes
  const int C3 = 1024;          // out_planes
  const int Ho = 14, Wo = 14;   // after stride 2

  // Workspace carve-out (256B aligned)
  char* ws = (char*)d_ws;
  size_t off = 0;
  auto carve = [&](size_t bytes) -> void* {
    void* p = ws + off;
    off = (off + bytes + 255) & ~(size_t)255;
    return p;
  };
  float*  h1  = (float*) carve((size_t)N * P  * H  * W  * 4);  // 51.4 MB
  float*  h2  = (float*) carve((size_t)N * P  * Ho * Wo * 4);  // 12.8 MB
  float*  scb = (float*) carve((size_t)N * C3 * Ho * Wo * 4);  // 51.4 MB
  bf16_t* wq1 = (bf16_t*)carve((size_t)P  * C1 * 2);
  bf16_t* wq2 = (bf16_t*)carve((size_t)P  * P * 9 * 2);
  bf16_t* wq3 = (bf16_t*)carve((size_t)C3 * P  * 2);
  bf16_t* wqd = (bf16_t*)carve((size_t)C3 * C1 * 2);
  float*  s1 = (float*)carve(C1 * 4); float* t1 = (float*)carve(C1 * 4);
  float*  s2 = (float*)carve(P  * 4); float* t2 = (float*)carve(P  * 4);
  float*  s3 = (float*)carve(P  * 4); float* t3 = (float*)carve(P  * 4);
  float*  sd = (float*)carve(C3 * 4); float* td = (float*)carve(C3 * 4);

  // 1) BN1 stats on x; ternarize all weights (independent)
  bn_stats_kernel<<<C1, 256, 0, stream>>>(x, bn1_g, bn1_b, s1, t1, N, C1, H * W);
  ternarize_kernel<<<P,  256, 0, stream>>>(w1,   wq1, C1);
  ternarize_kernel<<<P,  256, 0, stream>>>(w2,   wq2, P * 9);
  ternarize_kernel<<<C3, 256, 0, stream>>>(w3,   wq3, P);
  ternarize_kernel<<<C3, 256, 0, stream>>>(ds_w, wqd, C1);

  // 2) conv1: 1x1 s1 p0, 512->256, 28x28 (Npix 50176, K 512)
  conv_wmma_kernel<1, 1, 1, 0, 28, 28, true>
      <<<dim3((N * H * W) / 32, P / 128), 256, 0, stream>>>(
          x, s1, t1, wq1, h1, N, C1, H, W, P);

  // 3) BN2 stats on h1; conv2: 3x3 s2 p1, 256->256, 28x28 -> 14x14 (K 2304)
  bn_stats_kernel<<<P, 256, 0, stream>>>(h1, bn2_g, bn2_b, s2, t2, N, P, H * W);
  conv_wmma_kernel<3, 3, 2, 1, 14, 14, true>
      <<<dim3((N * Ho * Wo) / 32, P / 128), 256, 0, stream>>>(
          h1, s2, t2, wq2, h2, N, P, H, W, P);

  // 4) BN3 stats on h2; conv3: 1x1 s1 p0, 256->1024, 14x14 (K 256) -> d_out
  bn_stats_kernel<<<P, 256, 0, stream>>>(h2, bn3_g, bn3_b, s3, t3, N, P, Ho * Wo);
  conv_wmma_kernel<1, 1, 1, 0, 14, 14, true>
      <<<dim3((N * Ho * Wo) / 32, C3 / 128), 256, 0, stream>>>(
          h2, s3, t3, wq3, out, N, P, Ho, Wo, C3);

  // 5) shortcut: 1x1 stride-2 conv on RAW x, 512->1024 (K 512)
  conv_wmma_kernel<1, 1, 2, 0, 14, 14, false>
      <<<dim3((N * Ho * Wo) / 32, C3 / 128), 256, 0, stream>>>(
          x, nullptr, nullptr, wqd, scb, N, C1, H, W, C3);

  // 6) BN stats on shortcut output, then fused BN + residual add into d_out
  bn_stats_kernel<<<C3, 256, 0, stream>>>(scb, ds_g, ds_b, sd, td, N, C3, Ho * Wo);
  residual_add_kernel<<<N * C3, 256, 0, stream>>>(out, scb, sd, td, Ho * Wo);
}
